// AdaptiveSemanticFilter_15126874816924
// MI455X (gfx1250) — matrix operand order repaired and checked by reading, not compiled, not run on hardware
//
#include <hip/hip_runtime.h>
#include <hip/hip_bf16.h>

// Problem constants (from reference setup_inputs)
#define BATCH 256
#define L1C   512
#define L2C   256
#define DC    256
#define DU    (DC / 2)     // row length in packed bf16x2 dwords (128)
#define BM    128
#define BN    128
#define BK    32
#define NKSTAGES (DC / BK) // 8
#define LDS_STRIDE 20      // dwords per staged row: 16 data + 4 pad (16B aligned, bank spread)
#define NELEM (33554432u)  // BATCH*L1C*L2C

typedef __attribute__((ext_vector_type(16))) __bf16       v16bf;
typedef __attribute__((ext_vector_type(8)))  float        v8f;
typedef __attribute__((ext_vector_type(8)))  unsigned int u32x8;
typedef int v4i_vec __attribute__((vector_size(16)));   // matches builtin param type

// ---- CDNA5 async-copy + wait ----
#if __has_builtin(__builtin_amdgcn_s_wait_asynccnt)
#define WAIT_ASYNCCNT(N) __builtin_amdgcn_s_wait_asynccnt(N)
#else
#define WAIT_ASYNCCNT(N) asm volatile("s_wait_asynccnt %0" :: "i"(N) : "memory")
#endif

__device__ __forceinline__ void async_copy16(const unsigned* __restrict__ g,
                                             unsigned* l) {
    // global_load_async_to_lds_b128: (v4i AS1* src, v4i AS3* dst, imm off, imm cpol)
    __builtin_amdgcn_global_load_async_to_lds_b128(
        (__attribute__((address_space(1))) v4i_vec*)g,
        (__attribute__((address_space(3))) v4i_vec*)l,
        0, 0);
}

// round-to-nearest-even f32 -> bf16, packed pair into one dword (lo = first)
__device__ __forceinline__ unsigned pack_bf16x2(float lo, float hi) {
    unsigned a = __float_as_uint(lo);
    unsigned b = __float_as_uint(hi);
    a = (a + 0x7FFFu + ((a >> 16) & 1u)) >> 16;
    b = (b + 0x7FFFu + ((b >> 16) & 1u)) >> 16;
    return (a & 0xFFFFu) | (b << 16);
}

// gather a 16-half fragment (two aligned 16B LDS chunks) into v16bf
__device__ __forceinline__ v16bf load_frag(const unsigned* __restrict__ lds,
                                           int u0, int u1) {
    uint4 q0 = *reinterpret_cast<const uint4*>(lds + u0);
    uint4 q1 = *reinterpret_cast<const uint4*>(lds + u1);
    u32x8 r;
    r[0] = q0.x; r[1] = q0.y; r[2] = q0.z; r[3] = q0.w;
    r[4] = q1.x; r[5] = q1.y; r[6] = q1.z; r[7] = q1.w;
    return __builtin_bit_cast(v16bf, r);
}

// ---------------------------------------------------------------------------
// Kernel 1: fused row-norms + f32->bf16 pre-pack. One wave32 per 256-f32 row.
// rows [0, BATCH*L1C) -> visual (nv, Vb) ; rest -> textual (nt, Tb)
// ---------------------------------------------------------------------------
__global__ __launch_bounds__(256) void norms_convert_kernel(
        const float* __restrict__ V, const float* __restrict__ T,
        float* __restrict__ nv, float* __restrict__ nt,
        unsigned* __restrict__ Vb, unsigned* __restrict__ Tb) {
    const int RV  = BATCH * L1C;
    int gw   = (int)((blockIdx.x * 256u + threadIdx.x) >> 5);  // global wave = row id
    int lane = threadIdx.x & 31;
    bool isV = gw < RV;
    const float* src = isV ? (V + (size_t)gw * DC)
                           : (T + (size_t)(gw - RV) * DC);
    const float4* s4 = reinterpret_cast<const float4*>(src + lane * 8);
    float4 a = s4[0], b = s4[1];

    // packed bf16 copy (one-time conversion; GEMM hot loop stays conversion-free)
    uint4 q;
    q.x = pack_bf16x2(a.x, a.y);
    q.y = pack_bf16x2(a.z, a.w);
    q.z = pack_bf16x2(b.x, b.y);
    q.w = pack_bf16x2(b.z, b.w);
    unsigned* dst = (isV ? Vb + (size_t)gw * DU
                         : Tb + (size_t)(gw - RV) * DU) + lane * 4;
    *reinterpret_cast<uint4*>(dst) = q;

    float s = a.x*a.x + a.y*a.y + a.z*a.z + a.w*a.w
            + b.x*b.x + b.y*b.y + b.z*b.z + b.w*b.w;
#pragma unroll
    for (int m = 16; m > 0; m >>= 1) s += __shfl_xor(s, m, 32);
    if (lane == 0) {
        float nrm = sqrtf(s);
        if (isV) nv[gw] = nrm; else nt[gw - RV] = nrm;
    }
}

// ---------------------------------------------------------------------------
// Kernel 2: batched GEMM (bf16 WMMA, f32 accum) with double-buffered
//           GLOBAL_LOAD_ASYNC_TO_LDS_B128 panel staging + stats partials.
// Block = 256 thr = 8 waves; block tile 128x128; wave tile 32x64 (2x4 frags).
// ---------------------------------------------------------------------------
__global__ __launch_bounds__(256) void gemm_stats_kernel(
        const unsigned* __restrict__ Vb, const unsigned* __restrict__ Tb,
        float* __restrict__ out,
        const float* __restrict__ nv, const float* __restrict__ nt,
        float* __restrict__ partials) {
    __shared__ unsigned ldsA[2][BM * LDS_STRIDE];
    __shared__ unsigned ldsB[2][BN * LDS_STRIDE];
    __shared__ float    red[512];

    const int blk = blockIdx.x;
    const int b   = blk >> 3;          // 8 tiles per batch (4 M-tiles x 2 N-tiles)
    const int rem = blk & 7;
    const int m0  = (rem >> 1) * BM;
    const int n0  = (rem & 1)  * BN;

    const unsigned* Ab = Vb + (size_t)b * L1C * DU;
    const unsigned* Bb = Tb + (size_t)b * L2C * DU;

    const int tid  = threadIdx.x;
    const int lane = tid & 31;
    const int w    = tid >> 5;
    const int wm   = w >> 1;           // 0..3 : wave row (32 rows each)
    const int wn   = w & 1;            // 0..1 : wave col (64 cols each)
    const int hl   = lane >> 4;        // half-wave select
    const int l15  = lane & 15;

    // each thread async-copies 2 x 16B chunks per panel per stage (4 total)
    auto issue_stage = [&](int stage, int bufsel) {
#pragma unroll
        for (int c = 0; c < 2; ++c) {
            int ch   = 2 * tid + c;        // 0..511
            int row  = ch >> 2;            // 0..127
            int part = ch & 3;             // 16B chunk within the 64B row
            int loff = row * LDS_STRIDE + part * 4;
            size_t goff = (size_t)stage * 16 + part * 4;
            async_copy16(Ab + (size_t)(m0 + row) * DU + goff, &ldsA[bufsel][loff]);
            async_copy16(Bb + (size_t)(n0 + row) * DU + goff, &ldsB[bufsel][loff]);
        }
    };

    v8f acc[2][4];
#pragma unroll
    for (int i = 0; i < 2; ++i)
#pragma unroll
        for (int j = 0; j < 4; ++j) acc[i][j] = 0;

    issue_stage(0, 0);   // prologue: stage 0 -> buffer 0

    for (int kb = 0; kb < NKSTAGES; ++kb) {
        const int cur = kb & 1;
        if (kb + 1 < NKSTAGES) {
            issue_stage(kb + 1, cur ^ 1);  // overlap next panel copy with compute
            WAIT_ASYNCCNT(4);              // retire current stage's 4 copies (in-order)
        } else {
            WAIT_ASYNCCNT(0);
        }
        __syncthreads();                   // all waves' panel data visible

        const unsigned* lA = ldsA[cur];
        const unsigned* lB = ldsB[cur];

        // A 16x32: lane(l15,hl) row=l15, elem i -> k = (i<8?0:16) + 8*hl + (i&7)
        v16bf afrag[2];
#pragma unroll
        for (int ti = 0; ti < 2; ++ti) {
            int base = (wm * 32 + ti * 16 + l15) * LDS_STRIDE;
            afrag[ti] = load_frag(lA, base + 4 * hl, base + 8 + 4 * hl);
        }
        // B 32x16: lane(l15,hl) col=l15, elem i -> k = (i<8?0:16) + 8*hl + (i&7)
        v16bf bfrag[4];
#pragma unroll
        for (int tj = 0; tj < 4; ++tj) {
            int base = (wn * 64 + tj * 16 + l15) * LDS_STRIDE + 4 * hl;
            bfrag[tj] = load_frag(lB, base, base + 8);
        }

#pragma unroll
        for (int ti = 0; ti < 2; ++ti)
#pragma unroll
            for (int tj = 0; tj < 4; ++tj)
                acc[ti][tj] = __builtin_amdgcn_wmma_f32_16x16x32_bf16(
                    false, afrag[ti], false, bfrag[tj],
                    (short)0, acc[ti][tj], false, false);

        __syncthreads();                   // buffer free before next issue overwrites
    }

    // ---- epilogue: cosine-normalize, write sim, accumulate stats ----
    float s1 = 0.0f, s2 = 0.0f;
    float* outB = out + (size_t)b * L1C * L2C;
#pragma unroll
    for (int ti = 0; ti < 2; ++ti) {
#pragma unroll
        for (int tj = 0; tj < 4; ++tj) {
            int col   = n0 + wn * 64 + tj * 16 + l15;
            float ntc = nt[b * L2C + col];
#pragma unroll
            for (int r = 0; r < 8; ++r) {
                int   row = m0 + wm * 32 + ti * 16 + 8 * hl + r;
                float sim = acc[ti][tj][r] / (nv[b * L1C + row] * ntc + 1e-9f);
                s1 += sim;
                s2 += sim * sim;
                outB[(size_t)row * L2C + col] = sim;
            }
        }
    }

    red[tid]       = s1;
    red[256 + tid] = s2;
    __syncthreads();
#pragma unroll
    for (int s = 128; s > 0; s >>= 1) {
        if (tid < s) {
            red[tid]       += red[tid + s];
            red[256 + tid] += red[256 + tid + s];
        }
        __syncthreads();
    }
    if (tid == 0) {
        partials[2 * blk]     = red[0];
        partials[2 * blk + 1] = red[256];
    }
}

// ---------------------------------------------------------------------------
// Kernel 3: deterministic single-block reduce of 2048 partial pairs -> b2
// ---------------------------------------------------------------------------
__global__ __launch_bounds__(256) void reduce_thresh_kernel(
        const float* __restrict__ partials, float* __restrict__ threshOut) {
    __shared__ double rd[512];
    int tid = threadIdx.x;
    double s1 = 0.0, s2 = 0.0;
    for (int i = tid; i < 2048; i += 256) {
        s1 += (double)partials[2 * i];
        s2 += (double)partials[2 * i + 1];
    }
    rd[tid]       = s1;
    rd[256 + tid] = s2;
    __syncthreads();
#pragma unroll
    for (int s = 128; s > 0; s >>= 1) {
        if (tid < s) {
            rd[tid]       += rd[tid + s];
            rd[256 + tid] += rd[256 + tid + s];
        }
        __syncthreads();
    }
    if (tid == 0) {
        const double n  = (double)NELEM;
        double mu  = rd[0] / n;
        double var = (rd[256] - n * mu * mu) / (n - 1.0);
        if (var < 0.0) var = 0.0;
        float sigma = (float)sqrt(var);
        // b2 = mu2 + sigma2 * sqrt(-2*log(Z2 + EPS)), Z2 = 0.2
        *threshOut = (float)mu + sigma * sqrtf(-2.0f * logf(0.2f + 1e-9f));
    }
}

// ---------------------------------------------------------------------------
// Kernel 4: out = sim * (mask + EPS), mask = sim > thresh.  float4 in-place.
// ---------------------------------------------------------------------------
__global__ __launch_bounds__(256) void apply_mask_kernel(
        float* __restrict__ out, const float* __restrict__ thresh, int n4) {
    float t = *thresh;
    int i = (int)(blockIdx.x * 256u + threadIdx.x);
    if (i < n4) {
        float4* p = reinterpret_cast<float4*>(out);
        float4 v = p[i];
        v.x *= (v.x > t) ? (1.0f + 1e-9f) : 1e-9f;
        v.y *= (v.y > t) ? (1.0f + 1e-9f) : 1e-9f;
        v.z *= (v.z > t) ? (1.0f + 1e-9f) : 1e-9f;
        v.w *= (v.w > t) ? (1.0f + 1e-9f) : 1e-9f;
        p[i] = v;
    }
}

// ---------------------------------------------------------------------------
extern "C" void kernel_launch(void* const* d_in, const int* in_sizes, int n_in,
                              void* d_out, int out_size, void* d_ws, size_t ws_size,
                              hipStream_t stream) {
    const float* V = (const float*)d_in[0];   // visual_units  [256,512,256] f32
    const float* T = (const float*)d_in[1];   // textual_units [256,256,256] f32
    float* out = (float*)d_out;               // [256,512,256] f32

    // workspace layout (bytes)
    char* ws = (char*)d_ws;
    const size_t VB_BYTES = (size_t)BATCH * L1C * DU * 4;  // 64 MB packed bf16 V
    const size_t TB_BYTES = (size_t)BATCH * L2C * DU * 4;  // 32 MB packed bf16 T
    const size_t NV_BYTES = (size_t)BATCH * L1C * 4;       // 512 KB
    const size_t NT_BYTES = (size_t)BATCH * L2C * 4;       // 256 KB
    unsigned* Vb       = (unsigned*)ws;
    unsigned* Tb       = (unsigned*)(ws + VB_BYTES);
    float*    nv       = (float*)(ws + VB_BYTES + TB_BYTES);
    float*    nt       = (float*)(ws + VB_BYTES + TB_BYTES + NV_BYTES);
    float*    partials = (float*)(ws + VB_BYTES + TB_BYTES + NV_BYTES + NT_BYTES);
    float*    thresh   = (float*)(ws + VB_BYTES + TB_BYTES + NV_BYTES + NT_BYTES + 16384);

    // 196608 rows total, 8 wave-rows per 256-thread block
    norms_convert_kernel<<<24576, 256, 0, stream>>>(V, T, nv, nt, Vb, Tb);
    // 256 batches * 4 M-tiles * 2 N-tiles = 2048 blocks
    gemm_stats_kernel<<<2048, 256, 0, stream>>>(Vb, Tb, out, nv, nt, partials);
    reduce_thresh_kernel<<<1, 256, 0, stream>>>(partials, thresh);
    apply_mask_kernel<<<NELEM / 4 / 256, 256, 0, stream>>>(out, thresh, NELEM / 4);
}
